// DecoderRNN_31550829756940
// MI455X (gfx1250) — matrix operand order repaired and checked by reading, not compiled, not run on hardware
//
#include <hip/hip_runtime.h>

// ---------------------------------------------------------------------------
// CDNA5 (gfx1250) ConvLSTM stack: bf16x3 (truncated hi/lo split) WMMA implicit
// GEMM with async global->LDS weight staging and double-buffered LDS.
// ---------------------------------------------------------------------------

typedef __bf16 v16bf __attribute__((ext_vector_type(16)));
typedef float  v8f   __attribute__((ext_vector_type(8)));
typedef int    v4i   __attribute__((vector_size(16)));

union Frag16 {
    v16bf v;
    unsigned int u[8];
    unsigned short s[16];
};

#define BS_STRIDE 36   // ushort stride: 32 halves + 4 pad (72B rows)
#define AF_STRIDE 36   // float stride: 32 floats + 4 pad (144B rows, 16B aligned)

// Pack the high 16 bits of two f32 words into one dword: {hi(u1), hi(u0)}
#define PACK_HI2(u1, u0) __builtin_amdgcn_perm((u1), (u0), 0x07060302u)

// Async 16B global->LDS copy (tracked by ASYNCcnt). Builtin wants typed
// (v4i) pointers in addrspace(1)/(3); integer round-trips always compile.
#define ASYNC_CP16(gp, lp, off)                                                   \
    __builtin_amdgcn_global_load_async_to_lds_b128(                               \
        (__attribute__((address_space(1))) v4i*)(unsigned long long)              \
            (const void*)(gp),                                                    \
        (__attribute__((address_space(3))) v4i*)(unsigned int)                    \
            (unsigned long long)(void*)(lp),                                      \
        (off), 0)

// z[b, o, n] = sum_k W[o,k] * im2col(x_t,h)[k,n] + bias[o]
// K index k = ci*9 + ky*3 + kx over the concatenated (x: C ch, h: hid ch) input.
__global__ __launch_bounds__(128)
void convlstm_gates_gemm(const float* __restrict__ W4, const float* __restrict__ bias,
                         const float* __restrict__ xin, long long xBStride,
                         const float* __restrict__ hin, long long hBStride,
                         float* __restrict__ zout, long long zBStride,
                         int C, int hid, int HW, int Wd, int Hd)
{
    const int Ktot = (C + hid) * 9;            // always a multiple of 32 here
    const int n0 = blockIdx.x * 64;
    const int o0 = blockIdx.y * 64;
    const int b  = blockIdx.z;

    const float* xb = xin + (long long)b * xBStride;
    const float* hb = hin + (long long)b * hBStride;
    float*       zb = zout + (long long)b * zBStride;

    __shared__ float          Af32[2][64 * AF_STRIDE];   // raw f32 weight tiles
    __shared__ unsigned short BsH[2][64 * BS_STRIDE];    // bf16-hi im2col tiles
    __shared__ unsigned short BsL[2][64 * BS_STRIDE];    // bf16-lo im2col tiles

    const int tid  = threadIdx.x;
    const int wave = tid >> 5;
    const int lane = tid & 31;
    const int hsel = lane >> 4;     // half-wave select
    const int l16  = lane & 15;

    // A staging assignment: 64 rows x 32 floats; 2 threads per row, 64B each.
    const int rowA  = tid >> 1;
    const int ksegA = (tid & 1) << 4;
    const float* wrow = W4 + (long long)(o0 + rowA) * Ktot + ksegA;

    // B staging assignment: 64 pixels x 32 k; 2 threads per pixel.
    const int nB    = tid & 63;
    const int ksegB = (tid >> 6) << 4;
    const int nGlob = n0 + nB;
    const int py = nGlob / Wd;
    const int px = nGlob - py * Wd;
    const bool nValid = nGlob < HW;

    // ---- staging helpers -------------------------------------------------
    auto stageA = [&](int buf, int k0) {
        const float* g = wrow + k0;                       // 16B aligned
        float* l = &Af32[buf][rowA * AF_STRIDE + ksegA];  // 16B aligned
        ASYNC_CP16(g, l, 0);
        ASYNC_CP16(g, l, 16);
        ASYNC_CP16(g, l, 32);
        ASYNC_CP16(g, l, 48);
    };
    auto stageB = [&](int buf, int k0) {
        int kk = k0 + ksegB;
        int ci = kk / 9;
        int r  = kk - ci * 9;
        unsigned int* dh = (unsigned int*)&BsH[buf][nB * BS_STRIDE + ksegB];
        unsigned int* dl = (unsigned int*)&BsL[buf][nB * BS_STRIDE + ksegB];
#pragma unroll
        for (int jp = 0; jp < 8; ++jp) {
            unsigned int fu[2], rl[2];
#pragma unroll
            for (int e = 0; e < 2; ++e) {
                int ky = (r * 11) >> 5;        // r/3 for r in [0,8]
                int kx = r - ky * 3;
                int iy = py + ky - 1;
                int ix = px + kx - 1;
                float f = 0.f;
                if (nValid && (unsigned)iy < (unsigned)Hd && (unsigned)ix < (unsigned)Wd) {
                    int sp = iy * Wd + ix;
                    f = (ci < C) ? xb[(long long)ci * HW + sp]
                                 : hb[(long long)(ci - C) * HW + sp];
                }
                unsigned int u = __float_as_uint(f);
                fu[e] = u;                                         // hi = trunc
                rl[e] = __float_as_uint(f - __uint_as_float(u & 0xFFFF0000u));
                if (++r == 9) { r = 0; ++ci; }
            }
            dh[jp] = PACK_HI2(fu[1], fu[0]);
            dl[jp] = PACK_HI2(rl[1], rl[0]);
        }
    };

    const v8f vzero = {0.f, 0.f, 0.f, 0.f, 0.f, 0.f, 0.f, 0.f};
    v8f acc[4];
    acc[0] = vzero; acc[1] = vzero; acc[2] = vzero; acc[3] = vzero;

    // ---- prologue: stage tile 0 into buffer 0 ----------------------------
    stageA(0, 0);
    stageB(0, 0);
    __builtin_amdgcn_s_wait_asynccnt(0);
    __syncthreads();

    const int nsteps = Ktot >> 5;
    const int rowM = (wave << 4) + l16;

    for (int s = 0; s < nsteps; ++s) {
        const int cur = s & 1;
        const int nxt = cur ^ 1;

        // ---- prefetch next tile while current one computes ----
        if (s + 1 < nsteps) {
            const int k1 = (s + 1) << 5;
            stageA(nxt, k1);
            stageB(nxt, k1);
        }

        // ---- A fragment (16x32 bf16) built from raw f32 LDS tile:
        //      lanes 0-15 K{0..7,16..23}, lanes 16-31 K{8..15,24..31}
        Frag16 aH, aL;
        {
            const float* ap = &Af32[cur][rowM * AF_STRIDE + (hsel << 3)];
#pragma unroll
            for (int jp = 0; jp < 4; ++jp) {
                float f0 = ap[2 * jp];
                float f1 = ap[2 * jp + 1];
                float g0 = ap[2 * jp + 16];
                float g1 = ap[2 * jp + 17];
                unsigned int u0 = __float_as_uint(f0), u1 = __float_as_uint(f1);
                unsigned int w0 = __float_as_uint(g0), w1 = __float_as_uint(g1);
                aH.u[jp]     = PACK_HI2(u1, u0);
                aH.u[jp + 4] = PACK_HI2(w1, w0);
                unsigned int r0 = __float_as_uint(f0 - __uint_as_float(u0 & 0xFFFF0000u));
                unsigned int r1 = __float_as_uint(f1 - __uint_as_float(u1 & 0xFFFF0000u));
                unsigned int s0 = __float_as_uint(g0 - __uint_as_float(w0 & 0xFFFF0000u));
                unsigned int s1 = __float_as_uint(g1 - __uint_as_float(w1 & 0xFFFF0000u));
                aL.u[jp]     = PACK_HI2(r1, r0);
                aL.u[jp + 4] = PACK_HI2(s1, s0);
            }
        }

        // ---- 4 N-subtiles; B fragment (32x16 bf16): lanes 0-15 K{0..7,16..23},
        //      lanes 16-31 K{8..15,24..31}
#pragma unroll
        for (int nt = 0; nt < 4; ++nt) {
            const int col = (nt << 4) + l16;
            Frag16 bH, bL;
            const unsigned int* pb =
                (const unsigned int*)&BsH[cur][col * BS_STRIDE + (hsel << 3)];
            const unsigned int* qb =
                (const unsigned int*)&BsL[cur][col * BS_STRIDE + (hsel << 3)];
#pragma unroll
            for (int j = 0; j < 8; ++j) { bH.u[j] = pb[j + ((j >> 2) << 2)]; bL.u[j] = qb[j + ((j >> 2) << 2)]; }

            // bf16x3: hi*hi + hi*lo + lo*hi, f32 accumulate
            acc[nt] = __builtin_amdgcn_wmma_f32_16x16x32_bf16(false, aH.v, false, bH.v,
                                                              (short)0, acc[nt], false, false);
            acc[nt] = __builtin_amdgcn_wmma_f32_16x16x32_bf16(false, aH.v, false, bL.v,
                                                              (short)0, acc[nt], false, false);
            acc[nt] = __builtin_amdgcn_wmma_f32_16x16x32_bf16(false, aL.v, false, bH.v,
                                                              (short)0, acc[nt], false, false);
        }

        // next tile fully in LDS + everyone done reading `cur` before reuse
        __builtin_amdgcn_s_wait_asynccnt(0);
        __syncthreads();
    }

    // ---- epilogue: D layout => VGPR i: lanes 0-15 M=i, lanes 16-31 M=8+i ----
#pragma unroll
    for (int nt = 0; nt < 4; ++nt) {
        const int col = n0 + (nt << 4) + l16;
        if (col < HW) {
#pragma unroll
            for (int i = 0; i < 8; ++i) {
                int row = o0 + (wave << 4) + (hsel << 3) + i;
                zb[(long long)row * HW + col] = acc[nt][i] + bias[row];
            }
        }
    }
}

// LSTM cell update: c' = sig(f)*c + sig(i)*tanh(g); h' = sig(o)*tanh(c')
__global__ __launch_bounds__(256)
void convlstm_pointwise(const float* __restrict__ z, long long zBStride,
                        float* __restrict__ c_state, float* __restrict__ h_state,
                        float* __restrict__ dest, int hid, int HW, int T, int t, int doRelu)
{
    const int b = blockIdx.y;
    const long long plane = (long long)hid * HW;
    const long long idx = (long long)blockIdx.x * 256 + threadIdx.x;
    if (idx >= plane) return;

    const float* zbp = z + (long long)b * zBStride;
    float zi = zbp[idx];
    float zf = zbp[idx + plane];
    float zo = zbp[idx + 2 * plane];
    float zg = zbp[idx + 3 * plane];

    long long s = (long long)b * plane + idx;
    float c = c_state[s];

    float si = 1.f / (1.f + __expf(-zi));
    float sf = 1.f / (1.f + __expf(-zf));
    float so = 1.f / (1.f + __expf(-zo));
    float cn = sf * c + si * tanhf(zg);
    float hn = so * tanhf(cn);

    c_state[s] = cn;
    h_state[s] = hn;
    float o = doRelu ? fmaxf(hn, 0.f) : hn;
    dest[((long long)b * T + t) * plane + idx] = o;
}

// ---------------------------------------------------------------------------

static void run_stream(const float* x, int C0, int Hd, int Wd,
                       const float* const* wl, const float* const* bl, const int* hids,
                       float* hseq0, float* hseq1, float* out,
                       float* h_state, float* c_state, float* zbuf, hipStream_t stream)
{
    const int B = 2, T = 8;
    const int HW = Hd * Wd;
    for (int l = 0; l < 3; ++l) {
        const int hid = hids[l];
        const int Cl  = (l == 0) ? C0 : hids[l - 1];
        const float* src = (l == 0) ? x : (l == 1 ? hseq0 : hseq1);
        float* dst = (l == 0) ? hseq0 : (l == 1 ? hseq1 : out);

        (void)hipMemsetAsync(h_state, 0, (size_t)B * hid * HW * sizeof(float), stream);
        (void)hipMemsetAsync(c_state, 0, (size_t)B * hid * HW * sizeof(float), stream);

        dim3 gG((HW + 63) / 64, (4 * hid) / 64, B);
        dim3 bG(128);
        long long xBStride = (long long)T * Cl * HW;
        long long hBStride = (long long)hid * HW;
        long long zBStride = 4LL * hid * HW;
        long long plane = (long long)hid * HW;
        dim3 gP((unsigned)((plane + 255) / 256), B);

        for (int t = 0; t < T; ++t) {
            const float* xt = src + (long long)t * Cl * HW;
            hipLaunchKernelGGL(convlstm_gates_gemm, gG, bG, 0, stream,
                               wl[l], bl[l], xt, xBStride, h_state, hBStride,
                               zbuf, zBStride, Cl, hid, HW, Wd, Hd);
            hipLaunchKernelGGL(convlstm_pointwise, gP, dim3(256), 0, stream,
                               zbuf, zBStride, c_state, h_state, dst,
                               hid, HW, T, t, (l == 2) ? 1 : 0);
        }
    }
}

extern "C" void kernel_launch(void* const* d_in, const int* in_sizes, int n_in,
                              void* d_out, int out_size, void* d_ws, size_t ws_size,
                              hipStream_t stream)
{
    (void)in_sizes; (void)n_in; (void)out_size; (void)ws_size;

    const float* x23  = (const float*)d_in[0];
    const float* xfin = (const float*)d_in[1];
    const float* wA[3] = {(const float*)d_in[2], (const float*)d_in[4], (const float*)d_in[6]};
    const float* bA[3] = {(const float*)d_in[3], (const float*)d_in[5], (const float*)d_in[7]};
    const float* wB[3] = {(const float*)d_in[8], (const float*)d_in[10], (const float*)d_in[12]};
    const float* bB[3] = {(const float*)d_in[9], (const float*)d_in[11], (const float*)d_in[13]};

    // Workspace carve-up (floats)
    float* ws = (float*)d_ws;
    size_t off = 0;
    auto carve = [&](size_t n) { float* p = ws + off; off += n; return p; };
    float* zbuf    = carve(5914624);   // max gates buffer: B * 4*512 * 1444
    float* h_state = carve(1478656);   // max B * hid * HW
    float* c_state = carve(1478656);
    float* hseqA0  = carve(1478656);   // B*T*64*1444
    float* hseqA1  = carve(1478656);
    float* hseqB0  = carve(369664);    // B*T*64*361
    float* hseqB1  = carve(369664);

    float* out23 = (float*)d_out;
    float* outF  = out23 + (size_t)11829248;  // 2*8*512*38*38

    const int hidsA[3] = {64, 64, 512};
    const int hidsB[3] = {64, 64, 1024};

    run_stream(x23, 512, 38, 38, wA, bA, hidsA, hseqA0, hseqA1, out23,
               h_state, c_state, zbuf, stream);
    run_stream(xfin, 1024, 19, 19, wB, bB, hidsB, hseqB0, hseqB1, outF,
               h_state, c_state, zbuf, stream);
}
